// LePEAttention_30588757082820
// MI455X (gfx1250) — matrix operand order repaired and checked by reading, not compiled, not run on hardware
//
#include <hip/hip_runtime.h>

#define DIMC 128
#define HEAD_DIM 32
#define WIN 1024
#define KCH 128
#define NCHUNK (WIN / KCH)
// 32^-0.5 * log2(e): logits computed directly in log2 domain
#define SCALE_LOG2E 0.2550052662622225f

typedef __attribute__((ext_vector_type(16))) _Float16 v16h;
typedef __attribute__((ext_vector_type(8)))  _Float16 h8;
typedef __attribute__((ext_vector_type(8)))  float    v8f;

// window element p -> global L index (for window W-split index ww)
__device__ __forceinline__ int lmap(int p, int ww) {
  return ((p >> 6) << 8) | (((p >> 2) & 15) << 4) | (ww << 2) | (p & 3);
}

__device__ __forceinline__ v16h cat16(h8 a, h8 b) {
  return __builtin_shufflevector(a, b, 0,1,2,3,4,5,6,7,8,9,10,11,12,13,14,15);
}

__device__ __forceinline__ float fast_exp2(float x) {
#if __has_builtin(__builtin_amdgcn_exp2f)
  return __builtin_amdgcn_exp2f(x);
#else
  return exp2f(x);
#endif
}

__global__ __launch_bounds__(256)
void lepe_attn_kernel(const float* __restrict__ qkv, float* __restrict__ out) {
  __shared__ __align__(16) _Float16 Kt[2][KCH][HEAD_DIM];   // K chunks, row-major (double buffered)
  __shared__ __align__(16) _Float16 Vt[2][HEAD_DIM][KCH];   // V chunks, transposed
  __shared__ __align__(16) _Float16 Pt[8][16][32];          // per-wave P staging

  const int bid  = blockIdx.x;
  const int qc   = bid & 7;       // q chunk (128 rows)
  const int wh   = bid >> 3;      // 0..31
  const int head = wh & 3;
  const int w    = wh >> 2;       // window 0..7
  const int b    = w >> 2;        // batch
  const int ww   = w & 3;         // W-split index

  const int tid  = threadIdx.x;
  const int lane = tid & 31;
  const int wave = tid >> 5;
  const int half = lane >> 4;
  const int l16  = lane & 15;

  const float* qp = qkv;
  const float* kp = qkv + (size_t)2 * 4096 * 128;
  const float* vp = qkv + (size_t)4 * 4096 * 128;

  // ---- Q fragment (A-layout): 16 rows x 32 dims, pre-scaled into log2 domain ----
  const int q0 = qc * 128 + wave * 16;
  v16h qf;
  {
    const int gL = lmap(q0 + l16, ww);
    const float* row = qp + ((size_t)(b * 4096 + gL)) * DIMC + head * HEAD_DIM;
    const int o = half * 8;
    const float4* r0 = (const float4*)(row + o);
    const float4* r1 = (const float4*)(row + 16 + o);
    float4 a0 = r0[0], a1 = r0[1], c0 = r1[0], c1 = r1[1];
    float f[16] = {a0.x,a0.y,a0.z,a0.w, a1.x,a1.y,a1.z,a1.w,
                   c0.x,c0.y,c0.z,c0.w, c1.x,c1.y,c1.z,c1.w};
#pragma unroll
    for (int j = 0; j < 16; ++j) qf[j] = (_Float16)(f[j] * SCALE_LOG2E);
  }

  // constant ones B-fragment: row-sum reduction runs on the WMMA pipe
  v16h ones_h;
#pragma unroll
  for (int j = 0; j < 16; ++j) ones_h[j] = (_Float16)1.0f;

  // ---- staging helpers (double buffered, prefetch into registers) ----
  const int kl = tid >> 1;          // key row handled by this thread
  const int dh = (tid & 1) * 16;    // dim half
  float4 pkf[4], pvf[4];

  auto issue_loads = [&](int kc) {
    const int gL = lmap(kc * KCH + kl, ww);
    const size_t base = ((size_t)(b * 4096 + gL)) * DIMC + head * HEAD_DIM + dh;
    const float4* k4 = (const float4*)(kp + base);
    const float4* v4 = (const float4*)(vp + base);
    pkf[0] = k4[0]; pkf[1] = k4[1]; pkf[2] = k4[2]; pkf[3] = k4[3];
    pvf[0] = v4[0]; pvf[1] = v4[1]; pvf[2] = v4[2]; pvf[3] = v4[3];
  };

  auto commit = [&](int buf) {
    float kf[16] = {pkf[0].x,pkf[0].y,pkf[0].z,pkf[0].w, pkf[1].x,pkf[1].y,pkf[1].z,pkf[1].w,
                    pkf[2].x,pkf[2].y,pkf[2].z,pkf[2].w, pkf[3].x,pkf[3].y,pkf[3].z,pkf[3].w};
    float vf[16] = {pvf[0].x,pvf[0].y,pvf[0].z,pvf[0].w, pvf[1].x,pvf[1].y,pvf[1].z,pvf[1].w,
                    pvf[2].x,pvf[2].y,pvf[2].z,pvf[2].w, pvf[3].x,pvf[3].y,pvf[3].z,pvf[3].w};
    h8 k0, k1;
#pragma unroll
    for (int j = 0; j < 8; ++j) { k0[j] = (_Float16)kf[j]; k1[j] = (_Float16)kf[8 + j]; }
    *(h8*)&Kt[buf][kl][dh]     = k0;   // 2x ds_store_b128
    *(h8*)&Kt[buf][kl][dh + 8] = k1;
#pragma unroll
    for (int j = 0; j < 16; ++j) Vt[buf][dh + j][kl] = (_Float16)vf[j];  // transpose scatter
  };

  v8f o0 = {}, o1 = {};
  float mrow[8], lrow[8];
#pragma unroll
  for (int v = 0; v < 8; ++v) { mrow[v] = -INFINITY; lrow[v] = 0.f; }

  // prologue: fill buffer 0
  issue_loads(0);
  commit(0);
  __syncthreads();

  for (int kc = 0; kc < NCHUNK; ++kc) {
    const int buf = kc & 1;
    const bool pre = (kc + 1) < NCHUNK;
    if (pre) issue_loads(kc + 1);   // global loads in flight across the compute below

    // ---- S = Q K^T for all 128 keys: 8 back-to-back WMMAs ----
    v8f s[8];
#pragma unroll
    for (int sub = 0; sub < KCH / 32; ++sub) {
      const int kb = sub * 32;
      v16h bk0, bk1;
      {
        const h8* p0 = (const h8*)&Kt[buf][kb + l16][half * 8];
        const h8* p1 = (const h8*)&Kt[buf][kb + 16 + l16][half * 8];
        bk0 = cat16(p0[0], p0[2]);
        bk1 = cat16(p1[0], p1[2]);
      }
      v8f zero = {};
      s[2 * sub]     = __builtin_amdgcn_wmma_f32_16x16x32_f16(false, qf, false, bk0, (short)0, zero, false, false);
      s[2 * sub + 1] = __builtin_amdgcn_wmma_f32_16x16x32_f16(false, qf, false, bk1, (short)0, zero, false, false);
    }

    // ---- one online-softmax update per 128 keys ----
    float cm[8];
#pragma unroll
    for (int v = 0; v < 8; ++v) cm[v] = fmaxf(s[0][v], s[1][v]);
#pragma unroll
    for (int t = 2; t < 8; ++t)
#pragma unroll
      for (int v = 0; v < 8; ++v) cm[v] = fmaxf(cm[v], s[t][v]);
#pragma unroll
    for (int mask = 1; mask <= 8; mask <<= 1)
#pragma unroll
      for (int v = 0; v < 8; ++v) cm[v] = fmaxf(cm[v], __shfl_xor(cm[v], mask, 32));

    float alpha[8];
#pragma unroll
    for (int v = 0; v < 8; ++v) {
      float mn = fmaxf(mrow[v], cm[v]);
      alpha[v] = fast_exp2(mrow[v] - mn);
      mrow[v]  = mn;
    }
#pragma unroll
    for (int t = 0; t < 8; ++t)
#pragma unroll
      for (int v = 0; v < 8; ++v) s[t][v] = fast_exp2(s[t][v] - mrow[v]);

    // rescale accumulators once per chunk
#pragma unroll
    for (int v = 0; v < 8; ++v) { o0[v] *= alpha[v]; o1[v] *= alpha[v]; }

    // ---- P.V and P.ones (row sums chained through C) on the matrix pipe ----
    v8f rsf = {};
#pragma unroll
    for (int sub = 0; sub < KCH / 32; ++sub) {
      const int kb = sub * 32;
      // P: C-layout -> A-layout via per-wave LDS (same-wave, in-order)
#pragma unroll
      for (int v = 0; v < 8; ++v) {
        const int r = v + half * 8;
        Pt[wave][r][l16]      = (_Float16)s[2 * sub][v];
        Pt[wave][r][16 + l16] = (_Float16)s[2 * sub + 1][v];
      }
      v16h pf, bv0, bv1;
      {
        const int o = half * 8;
        const h8* pr0 = (const h8*)&Pt[wave][l16][o];
        const h8* pr1 = (const h8*)&Pt[wave][l16][16 + o];
        pf = cat16(pr0[0], pr1[0]);
        const h8* v0p = (const h8*)&Vt[buf][l16][kb + half * 8];
        const h8* v1p = (const h8*)&Vt[buf][16 + l16][kb + half * 8];
        bv0 = cat16(v0p[0], v0p[2]);
        bv1 = cat16(v1p[0], v1p[2]);
      }
      rsf = __builtin_amdgcn_wmma_f32_16x16x32_f16(false, pf, false, ones_h, (short)0, rsf, false, false);
      o0  = __builtin_amdgcn_wmma_f32_16x16x32_f16(false, pf, false, bv0,   (short)0, o0,  false, false);
      o1  = __builtin_amdgcn_wmma_f32_16x16x32_f16(false, pf, false, bv1,   (short)0, o1,  false, false);
    }
#pragma unroll
    for (int v = 0; v < 8; ++v) lrow[v] = lrow[v] * alpha[v] + rsf[v];

    if (pre) commit(buf ^ 1);   // convert + ds_store the prefetched chunk
    __syncthreads();            // one barrier per chunk: protects buf^1 fill and buf reuse
  }

  // ---- epilogue: normalize and scatter back through the window mapping ----
#pragma unroll
  for (int v = 0; v < 8; ++v) {
    const float inv = 1.0f / lrow[v];
    const int r  = v + half * 8;
    const int gL = lmap(q0 + r, ww);
    float* row = out + ((size_t)(b * 4096 + gL)) * DIMC + head * HEAD_DIM;
    row[l16]      = o0[v] * inv;
    row[16 + l16] = o1[v] * inv;
  }
}

extern "C" void kernel_launch(void* const* d_in, const int* in_sizes, int n_in,
                              void* d_out, int out_size, void* d_ws, size_t ws_size,
                              hipStream_t stream) {
  const float* qkv = (const float*)d_in[0];
  float* out = (float*)d_out;
  // 32 (window,head) pairs x 8 q-chunks of 128 rows; 8 waves/block, 16 q-rows/wave
  lepe_attn_kernel<<<dim3(256), dim3(256), 0, stream>>>(qkv, out);
}
